// OrthogonalBivectorBlock_32667521253926
// MI455X (gfx1250) — compile-verified, matrix operand order checked by reading, not run on hardware
//
#include <hip/hip_runtime.h>
#include <math.h>

#define PI_F 3.14159265358979323846f
#define LN_EPS 1e-5f

// problem constants
#define BB 2
#define LL 2048
#define DD 256
#define CC 32              // chunks along L
#define TT 64              // steps per chunk (CC*TT == LL)
#define NCOMP 104          // per-channel state floats: 16+4+16+16 complex

typedef __attribute__((ext_vector_type(16))) _Float16 v16h;
typedef __attribute__((ext_vector_type(8)))  float    v8f;

__device__ __forceinline__ float sigmoidf_(float x) { return 1.0f / (1.0f + expf(-x)); }
__device__ __forceinline__ float gelu_exact_(float x) {
    return 0.5f * x * (1.0f + erff(x * 0.70710678118654752f));
}

// ---------------------------------------------------------------------------
// WMMA GEMM: C[M,N] = act(A[M,K] @ W[K,N] + bias[N]) (+ resid[M,N])
// One wave per 16x32 C slab (two 16x16 WMMA tiles sharing the A fragment).
// K is a compile-time constant -> fully unrolled K loop, back-to-back WMMAs.
// A fragment (16-bit 16x32, wave32): lane = 16*half + m;
//   element j -> K = (j<8 ? j + 8*half : j + 8 + 8*half)  (two contiguous runs
//   of 8 -> two float4 pairs). B symmetric with n. C: c[i] -> row i+8*half.
// ---------------------------------------------------------------------------
template<int ACT, int KT>
__global__ void gemm_wmma_kernel(const float* __restrict__ A, const float* __restrict__ W,
                                 const float* __restrict__ bias, const float* __restrict__ resid,
                                 float* __restrict__ C, int M, int N) {
    const int tm = blockIdx.x * 16;
    const int tn = blockIdx.y * 32;          // two 16-wide N tiles per wave
    const int lane = threadIdx.x & 31;
    const int r    = lane & 15;
    const int half = lane >> 4;

    v8f acc0 = {}, acc1 = {};
    const float* arow = A + (size_t)(tm + r) * KT;
#pragma unroll
    for (int k0 = 0; k0 < KT; k0 += 32) {
        if (k0 + 32 < KT) {  // prefetch next K tile into WGP-level cache
            __builtin_prefetch(arow + k0 + 32 + 8 * half, 0, 3);
            __builtin_prefetch(&W[(size_t)(k0 + 32) * N + tn + r], 0, 3);
            __builtin_prefetch(&W[(size_t)(k0 + 32) * N + tn + 16 + r], 0, 3);
        }
        // A: two contiguous 8-float runs -> 4x float4 (global_load_b128)
        const float4* pa0 = reinterpret_cast<const float4*>(arow + k0 + 8 * half);
        const float4* pa1 = reinterpret_cast<const float4*>(arow + k0 + 16 + 8 * half);
        const float4 x0 = pa0[0], x1 = pa0[1], x2 = pa1[0], x3 = pa1[1];
        v16h a, b0, b1;
        a[0] = (_Float16)x0.x; a[1] = (_Float16)x0.y; a[2] = (_Float16)x0.z; a[3] = (_Float16)x0.w;
        a[4] = (_Float16)x1.x; a[5] = (_Float16)x1.y; a[6] = (_Float16)x1.z; a[7] = (_Float16)x1.w;
        a[8]  = (_Float16)x2.x; a[9]  = (_Float16)x2.y; a[10] = (_Float16)x2.z; a[11] = (_Float16)x2.w;
        a[12] = (_Float16)x3.x; a[13] = (_Float16)x3.y; a[14] = (_Float16)x3.z; a[15] = (_Float16)x3.w;
#pragma unroll
        for (int j = 0; j < 16; ++j) {
            const int kk = (j < 8) ? (j + 8 * half) : (j + 8 + 8 * half);
            const float* wrow = W + (size_t)(k0 + kk) * N;
            b0[j] = (_Float16)wrow[tn + r];
            b1[j] = (_Float16)wrow[tn + 16 + r];
        }
        acc0 = __builtin_amdgcn_wmma_f32_16x16x32_f16(
            false, a, false, b0, (short)0, acc0, false, false);
        acc1 = __builtin_amdgcn_wmma_f32_16x16x32_f16(
            false, a, false, b1, (short)0, acc1, false, false);
    }
#pragma unroll
    for (int i = 0; i < 8; ++i) {
        const int row  = tm + i + 8 * half;
        const int col0 = tn + r;
        const int col1 = tn + 16 + r;
        float v0 = acc0[i] + bias[col0];
        float v1 = acc1[i] + bias[col1];
        if (ACT == 1) { v0 = gelu_exact_(v0); v1 = gelu_exact_(v1); }
        if (resid) {
            v0 += resid[(size_t)row * N + col0];
            v1 += resid[(size_t)row * N + col1];
        }
        C[(size_t)row * N + col0] = v0;
        C[(size_t)row * N + col1] = v1;
    }
}

// ---------------------------------------------------------------------------
// Small heads
// ---------------------------------------------------------------------------
__global__ void phase16_kernel(const float* __restrict__ H, const float* __restrict__ W2,
                               const float* __restrict__ b2, float* __restrict__ PH,
                               int K, int rows) {
    const int idx = blockIdx.x * blockDim.x + threadIdx.x;
    if (idx >= rows * 16) return;
    const int row = idx >> 4, j = idx & 15;
    float s = b2[j];
    const float* h = H + (size_t)row * K;
    for (int k = 0; k < K; ++k) s += h[k] * W2[k * 16 + j];
    PH[idx] = tanhf(s) * PI_F;
}

__global__ void gate_kernel(const float* __restrict__ H, const float* __restrict__ W2,
                            const float* __restrict__ b2, float* __restrict__ G,
                            int K, int rows) {
    const int row = blockIdx.x * blockDim.x + threadIdx.x;
    if (row >= rows) return;
    float s = b2[0];
    const float* h = H + (size_t)row * K;
    for (int k = 0; k < K; ++k) s += h[k] * W2[k];
    G[row] = sigmoidf_(s);
}

// ---------------------------------------------------------------------------
// cos/sin tables: key/query/ltm phasors + joint (product-over-sets) phasors
// ---------------------------------------------------------------------------
__global__ void prep_cs_kernel(const float* __restrict__ KPH, const float* __restrict__ QPH,
                               const float* __restrict__ LPH,
                               float* __restrict__ KCS, float* __restrict__ QCS,
                               float* __restrict__ LCS,
                               float* __restrict__ JKCS, float* __restrict__ JQCS, int rows) {
    const int row = blockIdx.x * blockDim.x + threadIdx.x;
    if (row >= rows) return;
    float jk[4] = {0.f, 0.f, 0.f, 0.f}, jq[4] = {0.f, 0.f, 0.f, 0.f};
#pragma unroll
    for (int i = 0; i < 16; ++i) {
        const float kp = KPH[row * 16 + i];
        const float qp = QPH[row * 16 + i];
        const float lp = LPH[row * 16 + i];
        KCS[(size_t)(row * 16 + i) * 2 + 0] = cosf(kp);
        KCS[(size_t)(row * 16 + i) * 2 + 1] = sinf(kp);
        QCS[(size_t)(row * 16 + i) * 2 + 0] = cosf(qp);
        QCS[(size_t)(row * 16 + i) * 2 + 1] = sinf(qp);
        LCS[(size_t)(row * 16 + i) * 2 + 0] = cosf(lp);
        LCS[(size_t)(row * 16 + i) * 2 + 1] = sinf(lp);
        jk[i & 3] += kp;   // prod of unit phasors over sets == sum of phases
        jq[i & 3] += qp;
    }
#pragma unroll
    for (int p = 0; p < 4; ++p) {
        JKCS[(size_t)(row * 4 + p) * 2 + 0] = cosf(jk[p]);
        JKCS[(size_t)(row * 4 + p) * 2 + 1] = sinf(jk[p]);
        JQCS[(size_t)(row * 4 + p) * 2 + 0] = cosf(jq[p]);
        JQCS[(size_t)(row * 4 + p) * 2 + 1] = sinf(jq[p]);
    }
}

__global__ void postab_kernel(const float* __restrict__ freqs, float* __restrict__ T, int total) {
    const int idx = blockIdx.x * blockDim.x + threadIdx.x;
    if (idx >= total) return;
    const int t = idx & 15, l = idx >> 4;
    const float ang = (float)l * freqs[t] * (2.0f * PI_F);
    T[(size_t)idx * 2 + 0] = cosf(ang);
    T[(size_t)idx * 2 + 1] = sinf(ang);
}

// ---------------------------------------------------------------------------
// Write-gate prefix decomposition (phases only; independent of values)
// ---------------------------------------------------------------------------
__global__ void wg_p1_kernel(const float* __restrict__ JKCS, const float* __restrict__ LCS,
                             float* __restrict__ JCH, float* __restrict__ LCH) {
    const int idx = blockIdx.x * blockDim.x + threadIdx.x;   // b*CC + c
    if (idx >= BB * CC) return;
    const int b = idx / CC, c = idx % CC;
    float js[8] = {};
    float ls[32] = {};
    for (int l = c * TT; l < c * TT + TT; ++l) {
        const size_t row = (size_t)b * LL + l;
#pragma unroll
        for (int k = 0; k < 8; ++k)  js[k] += JKCS[row * 8 + k];
#pragma unroll
        for (int k = 0; k < 32; ++k) ls[k] += LCS[row * 32 + k];
    }
#pragma unroll
    for (int k = 0; k < 8; ++k)  JCH[(size_t)idx * 8 + k]  = js[k];
#pragma unroll
    for (int k = 0; k < 32; ++k) LCH[(size_t)idx * 32 + k] = ls[k];
}

__global__ void wg_p2_kernel(float* __restrict__ JCH, float* __restrict__ LCH) {
    const int idx = blockIdx.x * blockDim.x + threadIdx.x;   // BB * 40
    if (idx >= BB * 40) return;
    const int b = idx / 40;
    int k = idx % 40;
    if (k < 8) {
        float val = 0.f;
        for (int c = 0; c < CC; ++c) {
            const size_t a = (size_t)(b * CC + c) * 8 + k;
            const float tmp = JCH[a]; JCH[a] = val; val += tmp;
        }
    } else {
        k -= 8;
        float val = 0.f;
        for (int c = 0; c < CC; ++c) {
            const size_t a = (size_t)(b * CC + c) * 32 + k;
            const float tmp = LCH[a]; LCH[a] = val; val += tmp;
        }
    }
}

__global__ void wg_p3_kernel(const float* __restrict__ JKCS, const float* __restrict__ LCS,
                             const float* __restrict__ JCH, const float* __restrict__ LCH,
                             const float* __restrict__ ss_p, const float* __restrict__ sb_p,
                             float* __restrict__ WG, float* __restrict__ WG2) {
    const int idx = blockIdx.x * blockDim.x + threadIdx.x;   // b*CC + c
    if (idx >= BB * CC) return;
    const int b = idx / CC, c = idx % CC;
    const float ss = ss_p[0], sb = sb_p[0];
    float km_re[4], km_im[4], k2_re[16], k2_im[16];
#pragma unroll
    for (int p = 0; p < 4; ++p) { km_re[p] = JCH[(size_t)idx * 8 + 2 * p];
                                  km_im[p] = JCH[(size_t)idx * 8 + 2 * p + 1]; }
#pragma unroll
    for (int t = 0; t < 16; ++t) { k2_re[t] = LCH[(size_t)idx * 32 + 2 * t];
                                   k2_im[t] = LCH[(size_t)idx * 32 + 2 * t + 1]; }
    for (int l = c * TT; l < c * TT + TT; ++l) {
        const size_t row = (size_t)b * LL + l;
        const float pc = fmaxf((float)l, 1.0f);
        float rr = 0.f, ri = 0.f;
#pragma unroll
        for (int p = 0; p < 4; ++p) {
            const float jr = JKCS[(row * 4 + p) * 2 + 0];
            const float ji = JKCS[(row * 4 + p) * 2 + 1];
            rr += km_re[p] * jr + km_im[p] * ji;     // km_shift (exclusive) * conj(jk)
            ri += km_im[p] * jr - km_re[p] * ji;
            km_re[p] += jr; km_im[p] += ji;
        }
        const float fam = fminf(fmaxf(sqrtf(rr * rr + ri * ri) / (pc * 2.0f), 0.f), 1.f);
        WG[row] = sigmoidf_(ss * ((1.0f - fam) - 0.5f) + sb);
        float r2r = 0.f, r2i = 0.f;
#pragma unroll
        for (int t = 0; t < 16; ++t) {
            const float jr = LCS[(row * 16 + t) * 2 + 0];
            const float ji = LCS[(row * 16 + t) * 2 + 1];
            r2r += k2_re[t] * jr + k2_im[t] * ji;
            r2i += k2_im[t] * jr - k2_re[t] * ji;
            k2_re[t] += jr; k2_im[t] += ji;
        }
        const float fam2 = fminf(fmaxf(sqrtf(r2r * r2r + r2i * r2i) / (pc * 4.0f), 0.f), 1.f);
        WG2[row] = sigmoidf_(ss * ((1.0f - fam2) - 0.5f) + sb);
    }
}

// ---------------------------------------------------------------------------
// Value scan, chunked prefix decomposition. State comp layout (per b,c,d):
//   comps [0,32)  : bank    re/im pairs (i=0..15), bind coeff KCS, value v*wg
//   comps [32,40) : cross   re/im pairs (p=0..3),  bind coeff JKCS, value v*wg
//   comps [40,72) : pos     re/im pairs (t=0..15), bind coeff POSTAB, value v
//   comps [72,104): ltm     re/im pairs (t=0..15), bind coeff wg2*LCS, value lv
// CHS layout: [b][c][comp][d]  (d fastest -> coalesced)
// ---------------------------------------------------------------------------
__global__ void scan_p1_kernel(const float* __restrict__ KCS, const float* __restrict__ JKCS,
                               const float* __restrict__ LCS, const float* __restrict__ POSTAB,
                               const float* __restrict__ V, const float* __restrict__ LVAL,
                               const float* __restrict__ WG, const float* __restrict__ WG2,
                               float* __restrict__ CHS) {
    const int blk = blockIdx.x;                 // b*CC + c
    const int b = blk / CC, c = blk % CC;
    const int d = threadIdx.x;
    float st[NCOMP];
#pragma unroll
    for (int i = 0; i < NCOMP; ++i) st[i] = 0.f;
    for (int l = c * TT; l < c * TT + TT; ++l) {
        const size_t row = (size_t)b * LL + l;
        const float wg = WG[row], wg2 = WG2[row];
        const float v  = V[row * DD + d];
        const float lv = LVAL[row * DD + d];
        const float vg = v * wg;
#pragma unroll
        for (int i = 0; i < 16; ++i) {
            st[2 * i]     += KCS[(row * 16 + i) * 2 + 0] * vg;
            st[2 * i + 1] += KCS[(row * 16 + i) * 2 + 1] * vg;
        }
#pragma unroll
        for (int p = 0; p < 4; ++p) {
            st[32 + 2 * p]     += JKCS[(row * 4 + p) * 2 + 0] * vg;
            st[32 + 2 * p + 1] += JKCS[(row * 4 + p) * 2 + 1] * vg;
        }
#pragma unroll
        for (int t = 0; t < 16; ++t) {
            st[40 + 2 * t]     += POSTAB[((size_t)l * 16 + t) * 2 + 0] * v;
            st[40 + 2 * t + 1] += POSTAB[((size_t)l * 16 + t) * 2 + 1] * v;
        }
#pragma unroll
        for (int t = 0; t < 16; ++t) {
            st[72 + 2 * t]     += wg2 * LCS[(row * 16 + t) * 2 + 0] * lv;
            st[72 + 2 * t + 1] += wg2 * LCS[(row * 16 + t) * 2 + 1] * lv;
        }
    }
    const size_t base = ((size_t)blk * NCOMP) * DD + d;
#pragma unroll
    for (int i = 0; i < NCOMP; ++i) CHS[base + (size_t)i * DD] = st[i];
}

__global__ void scan_p2_kernel(float* __restrict__ CHS) {
    const int idx = blockIdx.x * blockDim.x + threadIdx.x;   // BB * NCOMP * DD
    if (idx >= BB * NCOMP * DD) return;
    const int b = idx / (NCOMP * DD);
    const int rem = idx % (NCOMP * DD);                      // comp*DD + d
    float val = 0.f;
    for (int c = 0; c < CC; ++c) {
        const size_t a = ((size_t)(b * CC + c) * NCOMP) * DD + rem;
        const float tmp = CHS[a]; CHS[a] = val; val += tmp;
    }
}

__global__ void scan_p3_kernel(const float* __restrict__ KCS, const float* __restrict__ QCS,
                               const float* __restrict__ JKCS, const float* __restrict__ JQCS,
                               const float* __restrict__ LCS, const float* __restrict__ POSTAB,
                               const float* __restrict__ V, const float* __restrict__ LVAL,
                               const float* __restrict__ WG, const float* __restrict__ WG2,
                               const float* __restrict__ GATE, const float* __restrict__ CHS,
                               const float* __restrict__ set_w, const float* __restrict__ pos_w,
                               const float* __restrict__ ltm_w, float* __restrict__ TOT) {
    const int blk = blockIdx.x;                 // b*CC + c
    const int b = blk / CC, c = blk % CC;
    const int d = threadIdx.x;

    float w[4];
    {
        float m = set_w[0];
        for (int s = 1; s < 4; ++s) m = fmaxf(m, set_w[s]);
        float den = 0.f;
        for (int s = 0; s < 4; ++s) { w[s] = expf(set_w[s] - m); den += w[s]; }
        for (int s = 0; s < 4; ++s) w[s] /= den;
    }
    const float posw = sigmoidf_(pos_w[0]);
    const float ltmw = sigmoidf_(ltm_w[0]);

    const size_t base = ((size_t)blk * NCOMP) * DD + d;
    float mre[16], mim[16], cre[4], cim[4], pre[16], pim[16], lre[16], lim[16];
#pragma unroll
    for (int i = 0; i < 16; ++i) { mre[i] = CHS[base + (size_t)(2 * i) * DD];
                                   mim[i] = CHS[base + (size_t)(2 * i + 1) * DD]; }
#pragma unroll
    for (int p = 0; p < 4; ++p)  { cre[p] = CHS[base + (size_t)(32 + 2 * p) * DD];
                                   cim[p] = CHS[base + (size_t)(32 + 2 * p + 1) * DD]; }
#pragma unroll
    for (int t = 0; t < 16; ++t) { pre[t] = CHS[base + (size_t)(40 + 2 * t) * DD];
                                   pim[t] = CHS[base + (size_t)(40 + 2 * t + 1) * DD]; }
#pragma unroll
    for (int t = 0; t < 16; ++t) { lre[t] = CHS[base + (size_t)(72 + 2 * t) * DD];
                                   lim[t] = CHS[base + (size_t)(72 + 2 * t + 1) * DD]; }

    for (int l = c * TT; l < c * TT + TT; ++l) {
        const size_t row = (size_t)b * LL + l;
        const float wg = WG[row], wg2 = WG2[row], g = GATE[row];
        const float v  = V[row * DD + d];
        const float lv = LVAL[row * DD + d];
        const float vg = v * wg;

        float bank = 0.f;
#pragma unroll
        for (int s = 0; s < 4; ++s) {
            float rs = 0.f;
#pragma unroll
            for (int p = 0; p < 4; ++p) {
                const int i = s * 4 + p;
                const float kc = KCS[(row * 16 + i) * 2 + 0];
                const float ks = KCS[(row * 16 + i) * 2 + 1];
                const float qc = QCS[(row * 16 + i) * 2 + 0];
                const float qs = QCS[(row * 16 + i) * 2 + 1];
                mre[i] += kc * vg; mim[i] += ks * vg;
                rs += mre[i] * qc + mim[i] * qs;       // Re(mem * conj(q))
            }
            bank += w[s] * rs;
        }
        float cb = 0.f;
#pragma unroll
        for (int p = 0; p < 4; ++p) {
            const float jkc = JKCS[(row * 4 + p) * 2 + 0];
            const float jks = JKCS[(row * 4 + p) * 2 + 1];
            const float jqc = JQCS[(row * 4 + p) * 2 + 0];
            const float jqs = JQCS[(row * 4 + p) * 2 + 1];
            cre[p] += jkc * vg; cim[p] += jks * vg;
            cb += cre[p] * jqc + cim[p] * jqs;
        }
        float tot = (bank + cb) * 0.2f;                // / (NS + 1)

        float pr = 0.f;
#pragma unroll
        for (int t = 0; t < 16; ++t) {
            const float pcc = POSTAB[((size_t)l * 16 + t) * 2 + 0];
            const float pss = POSTAB[((size_t)l * 16 + t) * 2 + 1];
            pre[t] += pcc * v; pim[t] += pss * v;      // ungated V
            pr += pre[t] * pcc + pim[t] * pss;
        }
        tot = g * tot + (1.0f - g) * (posw * pr);

        float lr = 0.f;
#pragma unroll
        for (int t = 0; t < 16; ++t) {
            const float lc = LCS[(row * 16 + t) * 2 + 0];
            const float ls = LCS[(row * 16 + t) * 2 + 1];
            lre[t] += wg2 * lc * lv; lim[t] += wg2 * ls * lv;
            lr += lre[t] * lc + lim[t] * ls;
        }
        lr *= rsqrtf((float)(l + 1) * 16.0f);          // / sqrt(pos1 * POS)
        tot += ltmw * lr;

        TOT[row * DD + d] = tot * rsqrtf((float)(l + 1) * 4.0f);  // / sqrt(pos1 * PP)
    }
}

// ---------------------------------------------------------------------------
// LayerNorm over D per row
// ---------------------------------------------------------------------------
__global__ void lnorm_kernel(const float* __restrict__ T, const float* __restrict__ gma,
                             const float* __restrict__ bta, float* __restrict__ OUT, int D) {
    const int row = blockIdx.x;
    const int d = threadIdx.x;
    __shared__ float red[256];
    const float t = T[(size_t)row * D + d];
    red[d] = t; __syncthreads();
    for (int s = 128; s > 0; s >>= 1) {
        if (d < s) red[d] += red[d + s];
        __syncthreads();
    }
    const float mu = red[0] / (float)D;
    __syncthreads();
    const float cv = t - mu;
    red[d] = cv * cv; __syncthreads();
    for (int s = 128; s > 0; s >>= 1) {
        if (d < s) red[d] += red[d + s];
        __syncthreads();
    }
    const float var = red[0] / (float)D;
    OUT[(size_t)row * D + d] = cv * rsqrtf(var + LN_EPS) * gma[d] + bta[d];
}

// ---------------------------------------------------------------------------
extern "C" void kernel_launch(void* const* d_in, const int* in_sizes, int n_in,
                              void* d_out, int out_size, void* d_ws, size_t ws_size,
                              hipStream_t stream) {
    const float* x   = (const float*)d_in[0];
    const float* kW1 = (const float*)d_in[1];  const float* kb1 = (const float*)d_in[2];
    const float* kW2 = (const float*)d_in[3];  const float* kb2 = (const float*)d_in[4];
    const float* qW1 = (const float*)d_in[5];  const float* qb1 = (const float*)d_in[6];
    const float* qW2 = (const float*)d_in[7];  const float* qb2 = (const float*)d_in[8];
    const float* vW  = (const float*)d_in[9];  const float* vb  = (const float*)d_in[10];
    const float* ln_g = (const float*)d_in[11]; const float* ln_b = (const float*)d_in[12];
    const float* oW  = (const float*)d_in[13]; const float* ob  = (const float*)d_in[14];
    const float* set_w = (const float*)d_in[15];
    const float* pos_freqs = (const float*)d_in[16];
    const float* pos_weight = (const float*)d_in[17];
    const float* gW1 = (const float*)d_in[18]; const float* gb1 = (const float*)d_in[19];
    const float* gW2 = (const float*)d_in[20]; const float* gb2 = (const float*)d_in[21];
    const float* lW1 = (const float*)d_in[22]; const float* lb1 = (const float*)d_in[23];
    const float* lW2 = (const float*)d_in[24]; const float* lb2 = (const float*)d_in[25];
    const float* lvW = (const float*)d_in[26]; const float* lvb = (const float*)d_in[27];
    const float* s_scale = (const float*)d_in[28];
    const float* s_bias  = (const float*)d_in[29];
    const float* ltm_w   = (const float*)d_in[30];

    float* out = (float*)d_out;
    const int R = BB * LL;   // 4096 rows

    // workspace layout (floats)
    float* ws = (float*)d_ws;
    float* H      = ws;                           // R*256 (hidden / tn buffer)
    float* Vb     = H      + (size_t)R * 256;
    float* LVAL   = Vb     + (size_t)R * 256;
    float* TOT    = LVAL   + (size_t)R * 256;
    float* KPH    = TOT    + (size_t)R * 256;     // R*16
    float* QPH    = KPH    + (size_t)R * 16;
    float* LPH    = QPH    + (size_t)R * 16;
    float* KCS    = LPH    + (size_t)R * 16;      // R*32
    float* QCS    = KCS    + (size_t)R * 32;
    float* LCS    = QCS    + (size_t)R * 32;
    float* JKCS   = LCS    + (size_t)R * 32;      // R*8
    float* JQCS   = JKCS   + (size_t)R * 8;
    float* WG     = JQCS   + (size_t)R * 8;       // R
    float* WG2    = WG     + (size_t)R;
    float* GATE   = WG2    + (size_t)R;
    float* POSTAB = GATE   + (size_t)R;           // L*32
    float* JCH    = POSTAB + (size_t)LL * 32;     // B*C*8
    float* LCH    = JCH    + (size_t)BB * CC * 8; // B*C*32
    float* CHS    = LCH    + (size_t)BB * CC * 32;// B*C*NCOMP*D (~6.8 MB)

    const dim3 wv(32);
    const dim3 g256(R / 16, 256 / 32);   // two 16-wide N tiles per wave
    const dim3 g128(R / 16, 128 / 32);

    // ---- feature GEMMs (WMMA) + small heads ----
    gemm_wmma_kernel<1, 256><<<g256, wv, 0, stream>>>(x, kW1, kb1, nullptr, H, R, 256);
    phase16_kernel<<<(R * 16 + 255) / 256, 256, 0, stream>>>(H, kW2, kb2, KPH, 256, R);
    gemm_wmma_kernel<1, 256><<<g256, wv, 0, stream>>>(x, qW1, qb1, nullptr, H, R, 256);
    phase16_kernel<<<(R * 16 + 255) / 256, 256, 0, stream>>>(H, qW2, qb2, QPH, 256, R);
    gemm_wmma_kernel<0, 256><<<g256, wv, 0, stream>>>(x, vW, vb, nullptr, Vb, R, 256);
    gemm_wmma_kernel<1, 256><<<g128, wv, 0, stream>>>(x, gW1, gb1, nullptr, H, R, 128);
    gate_kernel<<<(R + 255) / 256, 256, 0, stream>>>(H, gW2, gb2, GATE, 128, R);
    gemm_wmma_kernel<1, 256><<<g256, wv, 0, stream>>>(x, lW1, lb1, nullptr, H, R, 256);
    phase16_kernel<<<(R * 16 + 255) / 256, 256, 0, stream>>>(H, lW2, lb2, LPH, 256, R);
    gemm_wmma_kernel<0, 256><<<g256, wv, 0, stream>>>(x, lvW, lvb, nullptr, LVAL, R, 256);

    // ---- phasor tables ----
    prep_cs_kernel<<<(R + 255) / 256, 256, 0, stream>>>(KPH, QPH, LPH, KCS, QCS, LCS, JKCS, JQCS, R);
    postab_kernel<<<(LL * 16 + 255) / 256, 256, 0, stream>>>(pos_freqs, POSTAB, LL * 16);

    // ---- write gates: 3-pass parallel prefix over phases ----
    wg_p1_kernel<<<1, BB * CC, 0, stream>>>(JKCS, LCS, JCH, LCH);
    wg_p2_kernel<<<1, 128, 0, stream>>>(JCH, LCH);
    wg_p3_kernel<<<1, BB * CC, 0, stream>>>(JKCS, LCS, JCH, LCH, s_scale, s_bias, WG, WG2);

    // ---- value scan: 3-pass parallel prefix over chunks ----
    scan_p1_kernel<<<BB * CC, DD, 0, stream>>>(KCS, JKCS, LCS, POSTAB, Vb, LVAL, WG, WG2, CHS);
    scan_p2_kernel<<<(BB * NCOMP * DD + 255) / 256, 256, 0, stream>>>(CHS);
    scan_p3_kernel<<<BB * CC, DD, 0, stream>>>(KCS, QCS, JKCS, JQCS, LCS, POSTAB, Vb, LVAL,
                                               WG, WG2, GATE, CHS, set_w, pos_weight, ltm_w, TOT);

    // ---- LayerNorm + output projection with residual (WMMA) ----
    lnorm_kernel<<<R, DD, 0, stream>>>(TOT, ln_g, ln_b, H, DD);
    gemm_wmma_kernel<0, 256><<<g256, wv, 0, stream>>>(H, oW, ob, x, out, R, 256);
}
